// Head_90769838833936
// MI455X (gfx1250) — matrix operand order repaired and checked by reading, not compiled, not run on hardware
//
#include <hip/hip_runtime.h>

#define BATCH 1024
#define TSEQ  256
#define CDIM  128
#define HDIM  64
#define NWAVE 16
#define BLK   512

typedef __bf16 bf16_t;
typedef bf16_t v16bf __attribute__((ext_vector_type(16)));
typedef bf16_t v8bf  __attribute__((ext_vector_type(8)));
typedef bf16_t v4bf  __attribute__((ext_vector_type(4)));
typedef float  v8f   __attribute__((ext_vector_type(8)));

// ---- LDS layout (bytes) ----
#define OFF_XS   0          // x[b] as bf16 [256][128]            = 65536
#define OFF_WT   65536      // Wq^T,Wk^T,Wv^T bf16 3*[64][128]    = 49152
#define OFF_KS   114688     // K bf16 [256][64] row-major [s][h]  = 32768
#define OFF_VT   147456     // V^T bf16 [64][256]  [h][s]         = 32768
#define OFF_QS   180224     // Q bf16 [256][64], pre-scaled       = 32768
#define OFF_PR   212992     // per-wave P staging 16*[16][32]     = 16384
#define SMEM_BYTES 229376   // 224 KB < 320 KB/WGP (CDNA5)

__device__ __forceinline__ v16bf ld16(const bf16_t* p0, const bf16_t* p1) {
  v8bf lo = *(const v8bf*)p0;
  v8bf hi = *(const v8bf*)p1;
  return __builtin_shufflevector(lo, hi, 0,1,2,3,4,5,6,7,8,9,10,11,12,13,14,15);
}

// A-matrix 16x32 bf16 tile (row-major source, leading dim ld).
// lanes 0-15: M=lane, K=0..7 & 16..23 ; lanes 16-31: M=lane-16, K=8..15 & 24..31
__device__ __forceinline__ v16bf load_a(const bf16_t* base, int ld, int lane) {
  int half = lane >> 4, mr = lane & 15;
  const bf16_t* p = base + mr * ld + half * 8;
  return ld16(p, p + 16);
}

// B-matrix 32x16 bf16 tile; element (k,n) at base[n*ld + k] (K contiguous).
// lanes 0-15: N=lane, K=0..7 & 16..23 ; lanes 16-31: N=lane-16, K=8..15 & 24..31
__device__ __forceinline__ v16bf load_b(const bf16_t* base, int ld, int lane) {
  int half = lane >> 4, nr = lane & 15;
  const bf16_t* p = base + nr * ld + half * 8;
  return ld16(p, p + 16);
}

__device__ __forceinline__ v8f wmma_bf16(v16bf a, v16bf b, v8f c) {
  return __builtin_amdgcn_wmma_f32_16x16x32_bf16(false, a, false, b, (short)0, c,
                                                 false, false);
}

// reductions across 16-lane halves (wave32): xor masks 1..8 keep halves separate
__device__ __forceinline__ float rmax16(float v) {
  #pragma unroll
  for (int m = 1; m <= 8; m <<= 1) v = fmaxf(v, __shfl_xor(v, m, 32));
  return v;
}
__device__ __forceinline__ float rsum16(float v) {
  #pragma unroll
  for (int m = 1; m <= 8; m <<= 1) v += __shfl_xor(v, m, 32);
  return v;
}

// Projection: [256x128] @ [128x64] -> dst, specialized at COMPILE TIME on MAT so
// the store epilogue is straight-line (no exec-mask divergence on a uniform value).
// MAT: 0 = Q (scaled, row-major), 1 = K (row-major), 2 = V (transposed)
template <int MAT>
__device__ __forceinline__ void project(const bf16_t* __restrict__ xs,
                                        const bf16_t* __restrict__ wt,
                                        bf16_t* __restrict__ dst,
                                        int wave, int lane, float qscale) {
  const bf16_t* wbase = wt + MAT * HDIM * CDIM;
  for (int tile = wave; tile < 64; tile += NWAVE) {
    int mt = tile >> 2, nt = tile & 3;
    const bf16_t* wmat = wbase + (nt * 16) * CDIM;
    v8f acc = {};
    #pragma unroll
    for (int kk = 0; kk < 4; ++kk) {
      v16bf a  = load_a(xs + (mt * 16) * CDIM + kk * 32, CDIM, lane);
      v16bf bm = load_b(wmat + kk * 32, CDIM, lane);
      acc = wmma_bf16(a, bm, acc);
    }
    int col   = nt * 16 + (lane & 15);
    int rbase = mt * 16 + ((lane >> 4) << 3);
    #pragma unroll
    for (int r = 0; r < 8; ++r) {
      int row = rbase + r;
      if (MAT == 0)      dst[row * HDIM + col] = (bf16_t)(acc[r] * qscale);
      else if (MAT == 1) dst[row * HDIM + col] = (bf16_t)acc[r];
      else               dst[col * TSEQ + row] = (bf16_t)acc[r];
    }
  }
}

__global__ __launch_bounds__(BLK, 1)
void attn_head_kernel(const float* __restrict__ x,  const float* __restrict__ Wq,
                      const float* __restrict__ Wk, const float* __restrict__ Wv,
                      float* __restrict__ out) {
  extern __shared__ char smem[];
  bf16_t* xs    = (bf16_t*)(smem + OFF_XS);
  bf16_t* wt    = (bf16_t*)(smem + OFF_WT);
  bf16_t* ks    = (bf16_t*)(smem + OFF_KS);
  bf16_t* vt    = (bf16_t*)(smem + OFF_VT);
  bf16_t* qs    = (bf16_t*)(smem + OFF_QS);
  bf16_t* prall = (bf16_t*)(smem + OFF_PR);

  const int b    = blockIdx.x;
  const int tid  = threadIdx.x;
  const int lane = tid & 31;
  const int wave = tid >> 5;

  // ---- Phase 1: stage x[b] (f32->bf16) and W^T into LDS (single HBM pass) ----
  const float4* xg = (const float4*)(x + (size_t)b * TSEQ * CDIM);
  #pragma unroll 4
  for (int i = tid; i < TSEQ * CDIM / 4; i += BLK) {
    float4 f = xg[i];
    v4bf v; v[0] = (bf16_t)f.x; v[1] = (bf16_t)f.y; v[2] = (bf16_t)f.z; v[3] = (bf16_t)f.w;
    *(v4bf*)(xs + 4 * i) = v;
  }
  // W^T staging, unrolled per matrix (uniform pointers, no indirect array)
  for (int i = tid; i < CDIM * HDIM; i += BLK) {
    int c = i >> 6, h = i & 63;
    wt[0 * HDIM * CDIM + h * CDIM + c] = (bf16_t)Wq[i];
    wt[1 * HDIM * CDIM + h * CDIM + c] = (bf16_t)Wk[i];
    wt[2 * HDIM * CDIM + h * CDIM + c] = (bf16_t)Wv[i];
  }
  __syncthreads();

  // ---- Phase 2: Q,K,V projections via WMMA, compile-time specialized ----
  const float qscale = 0.08838834764831845f;     // C^-0.5
  project<0>(xs, wt, qs, wave, lane, qscale);
  project<1>(xs, wt, ks, wave, lane, qscale);
  project<2>(xs, wt, vt, wave, lane, qscale);
  __syncthreads();

  // ---- Phase 3: causal flash attention; wave w owns query rows [16w,16w+16) ----
  const int qt = wave;
  bf16_t* pr = prall + wave * (16 * 32);
  v16bf qa0 = load_a(qs + (qt * 16) * HDIM + 0,  HDIM, lane);
  v16bf qa1 = load_a(qs + (qt * 16) * HDIM + 32, HDIM, lane);

  float run_max[8], run_sum[8];
  #pragma unroll
  for (int r = 0; r < 8; ++r) { run_max[r] = -INFINITY; run_sum[r] = 0.f; }
  v8f o0 = {}, o1 = {}, o2 = {}, o3 = {};

  const int nblk  = ((qt + 1) * 16 + 31) >> 5;   // 32-key blocks covering keys<=qmax
  const int nr    = lane & 15;
  const int rbase = (lane >> 4) << 3;

  for (int kb = 0; kb < nblk; ++kb) {
    // S = Q * K^T for 32 keys (two 16-col tiles)
    v8f s0 = {}, s1 = {};
    const bf16_t* kb0 = ks + (kb * 32) * HDIM;
    const bf16_t* kb1 = ks + (kb * 32 + 16) * HDIM;
    s0 = wmma_bf16(qa0, load_b(kb0 + 0,  HDIM, lane), s0);
    s0 = wmma_bf16(qa1, load_b(kb0 + 32, HDIM, lane), s0);
    s1 = wmma_bf16(qa0, load_b(kb1 + 0,  HDIM, lane), s1);
    s1 = wmma_bf16(qa1, load_b(kb1 + 32, HDIM, lane), s1);

    if (kb == nblk - 1) {                        // causal mask, diagonal block only
      #pragma unroll
      for (int r = 0; r < 8; ++r) {
        int qrow = qt * 16 + rbase + r;
        if (kb * 32 + nr      > qrow) s0[r] = -INFINITY;
        if (kb * 32 + 16 + nr > qrow) s1[r] = -INFINITY;
      }
    }

    // online softmax per row (rows live in 16-lane halves, 8 rows per lane-set)
    float alpha[8];
    #pragma unroll
    for (int r = 0; r < 8; ++r) {
      float bmax = rmax16(fmaxf(s0[r], s1[r]));
      float mnew = fmaxf(run_max[r], bmax);
      float p0 = __expf(s0[r] - mnew);
      float p1 = __expf(s1[r] - mnew);
      alpha[r] = __expf(run_max[r] - mnew);
      run_sum[r] = run_sum[r] * alpha[r] + rsum16(p0 + p1);
      run_max[r] = mnew;
      int row = rbase + r;
      pr[row * 32 + nr]      = (bf16_t)p0;       // stage P row-major [16][32]
      pr[row * 32 + 16 + nr] = (bf16_t)p1;
    }
    asm volatile("s_wait_dscnt 0" ::: "memory"); // cross-lane P visible before A-load

    #pragma unroll
    for (int r = 0; r < 8; ++r) {
      o0[r] *= alpha[r]; o1[r] *= alpha[r]; o2[r] *= alpha[r]; o3[r] *= alpha[r];
    }
    v16bf pa = load_a(pr, 32, lane);
    o0 = wmma_bf16(pa, load_b(vt + 0 * 16 * TSEQ + kb * 32, TSEQ, lane), o0);
    o1 = wmma_bf16(pa, load_b(vt + 1 * 16 * TSEQ + kb * 32, TSEQ, lane), o1);
    o2 = wmma_bf16(pa, load_b(vt + 2 * 16 * TSEQ + kb * 32, TSEQ, lane), o2);
    o3 = wmma_bf16(pa, load_b(vt + 3 * 16 * TSEQ + kb * 32, TSEQ, lane), o3);
    asm volatile("s_wait_dscnt 0" ::: "memory"); // P reads done before next overwrite
  }

  // ---- epilogue: normalize and store f32 output ----
  float* og = out + (size_t)b * TSEQ * HDIM + (qt * 16) * HDIM;
  #pragma unroll
  for (int r = 0; r < 8; ++r) {
    float inv = 1.0f / run_sum[r];
    int row = rbase + r;
    og[row * HDIM +  0 + nr] = o0[r] * inv;
    og[row * HDIM + 16 + nr] = o1[r] * inv;
    og[row * HDIM + 32 + nr] = o2[r] * inv;
    og[row * HDIM + 48 + nr] = o3[r] * inv;
  }
}

extern "C" void kernel_launch(void* const* d_in, const int* in_sizes, int n_in,
                              void* d_out, int out_size, void* d_ws, size_t ws_size,
                              hipStream_t stream) {
  const float* x  = (const float*)d_in[0];
  const float* Wq = (const float*)d_in[1];
  const float* Wk = (const float*)d_in[2];
  const float* Wv = (const float*)d_in[3];
  float* out = (float*)d_out;
  (void)in_sizes; (void)n_in; (void)out_size; (void)d_ws; (void)ws_size;

  hipFuncSetAttribute((const void*)attn_head_kernel,
                      hipFuncAttributeMaxDynamicSharedMemorySize, SMEM_BYTES);
  attn_head_kernel<<<dim3(BATCH), dim3(BLK), SMEM_BYTES, stream>>>(x, Wq, Wk, Wv, out);
}
